// MPAdaptive_89524298317937
// MI455X (gfx1250) — compile-verified, matrix operand order checked by reading, not run on hardware
//
#include <hip/hip_runtime.h>
#include <math.h>

#define DEVI __device__ __forceinline__

typedef __attribute__((ext_vector_type(16))) __bf16 v16bf;
typedef __attribute__((ext_vector_type(4)))  __bf16 v4bf;
typedef __attribute__((ext_vector_type(8)))  float  v8f;

// Problem constants
constexpr int Bb  = 64;
constexpr int Nn  = 512;
constexpr int Hh  = 128;
constexpr int Ff  = 9;
constexpr int BN  = Bb * Nn;      // 32768 rows
constexpr int XHK = 288;          // padded cat(message,jets,h) K: 128+9+128 -> 288
constexpr int WHXK = 160;         // padded Wh K: 137 -> 160
constexpr int NT   = Hh / 16;     // 8 column tiles for all weight matrices
constexpr int SBLD = 520;         // padded bf16 LDS row stride (bank spread)

// ---------------- fast math ----------------

DEVI float fast_tanh(float x) {
#if __has_builtin(__builtin_amdgcn_tanhf)
  return __builtin_amdgcn_tanhf(x);
#elif __has_builtin(__builtin_amdgcn_tanh_f32)
  return __builtin_amdgcn_tanh_f32(x);
#else
  return tanhf(x);
#endif
}
DEVI float sigmoidf_(float x) { return 1.0f / (1.0f + __expf(-x)); }

// ---------------- WMMA helpers ----------------

DEVI v8f wmma_bf16(v16bf a, v16bf b, v8f c) {
  return __builtin_amdgcn_wmma_f32_16x16x32_bf16(
      /*neg_a=*/false, a, /*neg_b=*/false, b,
      /*c_mod=*/(short)0, c, /*reuse_a=*/false, /*reuse_b=*/false);
}

// A-fragment: 16x32 bf16 tile, row-major. Lane l<16: row l, K 0..7 & 16..23;
// lanes 16..31: row l, K 8..15 & 24..31 (ISA 7.12.2 16-bit A layout).
DEVI v16bf frag_rows_bf(const __bf16* p0, int ld) {
  int lane = threadIdx.x & 31;
  int l = lane & 15, hi = lane >> 4;
  const __bf16* p = p0 + (size_t)l * ld + hi * 8;
  v16bf f;
#pragma unroll
  for (int j = 0; j < 8; ++j) { f[j] = p[j]; f[8 + j] = p[16 + j]; }
  return f;
}

// Packed B-fragment: weights pre-swizzled so each lane's 16 bf16 are contiguous.
DEVI v16bf frag_packed(const __bf16* tile_base) {
  int lane = threadIdx.x & 31;
  return *(const v16bf*)(tile_base + (size_t)lane * 16);
}

// ---------------- Prep: pack cat(W;U;0) [totalK x 128] into B-fragment layout ----
__global__ void k_pack_cat(__bf16* __restrict__ dst, const float* __restrict__ W, int rowsW,
                           const float* __restrict__ U, int rowsU, int totalK) {
  int idx = blockIdx.x * blockDim.x + threadIdx.x;
  int total = totalK * Hh;
  if (idx >= total) return;
  int j    = idx & 15;
  int lane = (idx >> 4) & 31;
  int tile = idx >> 9;
  int nt = tile & (NT - 1);
  int kk = tile / NT;
  int l = lane & 15, hi = lane >> 4;
  int n  = nt * 16 + l;
  int ko = (j < 8) ? (hi * 8 + j) : (16 + hi * 8 + (j - 8));
  int k  = kk * 32 + ko;
  float v = 0.0f;
  if (k < rowsW)              v = W[k * Hh + n];
  else if (k < rowsW + rowsU) v = U[(k - rowsW) * Hh + n];
  dst[idx] = (__bf16)v;
}

// Build xh[BN,288]: [128..136]=jets, [137..264]=h (mirrored into h_bf), [265..287]=0.
// Cols 0..127 (message) are fully written later by k_attn.
__global__ void k_prep_xh(const float* __restrict__ h, const float* __restrict__ jets,
                          __bf16* __restrict__ xh, __bf16* __restrict__ h_bf) {
  size_t idx = (size_t)blockIdx.x * blockDim.x + threadIdx.x;
  size_t total = (size_t)BN * (XHK - 128);
  if (idx >= total) return;
  int bn = (int)(idx / (XHK - 128));
  int c  = 128 + (int)(idx - (size_t)bn * (XHK - 128));
  __bf16 v = (__bf16)0.0f;
  if (c < 128 + Ff) {
    v = (__bf16)jets[(size_t)bn * Ff + (c - 128)];
  } else if (c < 137 + Hh) {
    float hv = h[(size_t)bn * Hh + (c - 137)];
    v = (__bf16)hv;
    h_bf[(size_t)bn * Hh + (c - 137)] = v;
  }
  xh[(size_t)bn * XHK + c] = v;
}

// ---------------- z/m GEMM: z=bf16(h@W_adj+b) row-major, mT=bf16(h@W_msg+b) transposed ---
__global__ void k_zm(const __bf16* __restrict__ h_bf,
                     const __bf16* __restrict__ WadjP, const __bf16* __restrict__ WmsgP,
                     const float* __restrict__ badj, const float* __restrict__ bmsg,
                     __bf16* __restrict__ z_bf, __bf16* __restrict__ mT) {
  int r0 = blockIdx.x * 16;
  int w  = threadIdx.x >> 5;
  int n0 = w * 16;
  int lane = threadIdx.x & 31;
  int l = lane & 15, hi = lane >> 4;
  v8f az = {}, am = {};
#pragma unroll
  for (int kk = 0; kk < Hh / 32; ++kk) {
    v16bf a  = frag_rows_bf(h_bf + (size_t)r0 * Hh + kk * 32, Hh);
    v16bf bz = frag_packed(WadjP + (size_t)(kk * NT + w) * 512);
    v16bf bm = frag_packed(WmsgP + (size_t)(kk * NT + w) * 512);
    az = wmma_bf16(a, bz, az);
    am = wmma_bf16(a, bm, am);
  }
  int b    = r0 >> 9;
  int kloc = r0 & 511;
#pragma unroll
  for (int r = 0; r < 8; ++r) {
    int m = hi * 8 + r;
    int ng = n0 + l;
    z_bf[(size_t)(r0 + m) * Hh + ng] = (__bf16)(az[r] + badj[ng]);
    mT[((size_t)b * Hh + ng) * Nn + (kloc + m)] = (__bf16)(am[r] + bmsg[ng]);
  }
}

// ---------------- Attention: scores -> softmax -> A out -> message ----------------
// grid.x = B * (N/16); block = 256 (8 waves).
// LDS: f32 scores 16x512 (32KB) + bf16 A copy 16x520 (16.3KB)
__global__ void k_attn(const __bf16* __restrict__ z_bf, const __bf16* __restrict__ mT,
                       const float* __restrict__ mask,
                       float* __restrict__ A_out, __bf16* __restrict__ xh) {
  __shared__ float  s[16 * Nn];
  __shared__ __bf16 sb[16 * SBLD];
  int b  = blockIdx.x >> 5;
  int rt = blockIdx.x & 31;
  int r0 = rt * 16;
  int w  = threadIdx.x >> 5;
  int lane = threadIdx.x & 31;
  int l = lane & 15, hi = lane >> 4;
  const __bf16* zb = z_bf + (size_t)b * Nn * Hh;

  // --- scores = z_tile @ z_b^T  (B column n = z row n: contiguous row loads) ---
  v16bf afr[Hh / 32];
#pragma unroll
  for (int kk = 0; kk < Hh / 32; ++kk)
    afr[kk] = frag_rows_bf(zb + (size_t)r0 * Hh + kk * 32, Hh);
#pragma unroll
  for (int t = 0; t < 4; ++t) {
    int ct = w + t * 8;
    v8f acc = {};
#pragma unroll
    for (int kk = 0; kk < Hh / 32; ++kk) {
      v16bf bf = frag_rows_bf(zb + (size_t)(ct * 16) * Hh + kk * 32, Hh);
      acc = wmma_bf16(afr[kk], bf, acc);
    }
#pragma unroll
    for (int r = 0; r < 8; ++r) {
      int m = hi * 8 + r;
      s[m * Nn + ct * 16 + l] = acc[r];
    }
  }
  __syncthreads();

  // --- masked row softmax, fully register-resident (2 rows/wave, 4xfloat4/lane/row) ---
  float4 mk[4];
#pragma unroll
  for (int q = 0; q < 4; ++q)
    mk[q] = *(const float4*)&mask[(size_t)b * Nn + q * 128 + lane * 4];

#pragma unroll
  for (int rr = 0; rr < 2; ++rr) {
    int r = w * 2 + rr;
    float4 v[4];
    float mx = -3.4e38f;
#pragma unroll
    for (int q = 0; q < 4; ++q) {
      float4 t = *(const float4*)&s[r * Nn + q * 128 + lane * 4];
      t.x = mk[q].x > 0.0f ? t.x : -1e9f;
      t.y = mk[q].y > 0.0f ? t.y : -1e9f;
      t.z = mk[q].z > 0.0f ? t.z : -1e9f;
      t.w = mk[q].w > 0.0f ? t.w : -1e9f;
      v[q] = t;
      mx = fmaxf(mx, fmaxf(fmaxf(t.x, t.y), fmaxf(t.z, t.w)));
    }
#pragma unroll
    for (int off = 16; off > 0; off >>= 1) mx = fmaxf(mx, __shfl_xor(mx, off, 32));
    float sum = 0.0f;
#pragma unroll
    for (int q = 0; q < 4; ++q) {
      v[q].x = __expf(v[q].x - mx);
      v[q].y = __expf(v[q].y - mx);
      v[q].z = __expf(v[q].z - mx);
      v[q].w = __expf(v[q].w - mx);
      sum += (v[q].x + v[q].y) + (v[q].z + v[q].w);
    }
#pragma unroll
    for (int off = 16; off > 0; off >>= 1) sum += __shfl_xor(sum, off, 32);
    float scl = (1.0f / sum) * mask[(size_t)b * Nn + r0 + r];
    float* Arow = A_out + ((size_t)b * Nn + r0 + r) * Nn;
#pragma unroll
    for (int q = 0; q < 4; ++q) {
      float4 p;
      p.x = v[q].x * scl; p.y = v[q].y * scl; p.z = v[q].z * scl; p.w = v[q].w * scl;
      *(float4*)&Arow[q * 128 + lane * 4] = p;
      v4bf pb;
      pb[0] = (__bf16)p.x; pb[1] = (__bf16)p.y; pb[2] = (__bf16)p.z; pb[3] = (__bf16)p.w;
      *(v4bf*)&sb[r * SBLD + q * 128 + lane * 4] = pb;
    }
  }
  __syncthreads();

  // --- message = tanh(A_tile @ m_b): A-frag from bf16 LDS, B-frag from mT rows ---
  int n0 = w * 16;
  const __bf16* mtb = mT + (size_t)b * Hh * Nn;
  v8f acc = {};
#pragma unroll
  for (int kk = 0; kk < Nn / 32; ++kk) {
    v16bf a;
    {
      const __bf16* p = &sb[l * SBLD + kk * 32 + hi * 8];
#pragma unroll
      for (int j = 0; j < 8; ++j) { a[j] = p[j]; a[8 + j] = p[16 + j]; }
    }
    v16bf bf = frag_rows_bf(mtb + (size_t)n0 * Nn + kk * 32, Nn);
    acc = wmma_bf16(a, bf, acc);
  }
#pragma unroll
  for (int r = 0; r < 8; ++r) {
    int m = hi * 8 + r;
    int row = b * Nn + r0 + m;
    xh[(size_t)row * XHK + n0 + l] = (__bf16)fast_tanh(acc[r]);
  }
}

// ---------------- GRU gates: zg, rg (fused), rgh = bf16(rg*h) ----------------
__global__ void k_gates(const __bf16* __restrict__ xh,
                        const __bf16* __restrict__ WzP, const __bf16* __restrict__ WrP,
                        const float* __restrict__ bz, const float* __restrict__ br,
                        const float* __restrict__ h,
                        float* __restrict__ zg, __bf16* __restrict__ rgh) {
  int r0 = blockIdx.x * 16;
  int w  = threadIdx.x >> 5;
  int n0 = w * 16;
  int lane = threadIdx.x & 31;
  int l = lane & 15, hi = lane >> 4;
  v8f accz = {}, accr = {};
#pragma unroll
  for (int kk = 0; kk < XHK / 32; ++kk) {
    v16bf a   = frag_rows_bf(xh + (size_t)r0 * XHK + kk * 32, XHK);
    v16bf bzf = frag_packed(WzP + (size_t)(kk * NT + w) * 512);
    v16bf brf = frag_packed(WrP + (size_t)(kk * NT + w) * 512);
    accz = wmma_bf16(a, bzf, accz);
    accr = wmma_bf16(a, brf, accr);
  }
#pragma unroll
  for (int r = 0; r < 8; ++r) {
    int m = hi * 8 + r;
    int ng = n0 + l;
    size_t o = (size_t)(r0 + m) * Hh + ng;
    float zv = sigmoidf_(accz[r] + bz[ng]);
    float rv = sigmoidf_(accr[r] + br[ng]);
    zg[o]  = zv;
    rgh[o] = (__bf16)(rv * h[o]);
  }
}

// ---------------- Final: h_tilde = tanh(x@Wh + (rg*h)@Uh + bh); GRU blend ----------------
__global__ void k_final(const __bf16* __restrict__ xh,  const __bf16* __restrict__ WhP,
                        const __bf16* __restrict__ rgh, const __bf16* __restrict__ UhP,
                        const float* __restrict__ bh, const float* __restrict__ h,
                        const float* __restrict__ zg, float* __restrict__ hnew) {
  int r0 = blockIdx.x * 16;
  int w  = threadIdx.x >> 5;
  int n0 = w * 16;
  int lane = threadIdx.x & 31;
  int l = lane & 15, hi = lane >> 4;
  v8f acc = {};
#pragma unroll
  for (int kk = 0; kk < WHXK / 32; ++kk) {       // x @ Wh (padded K=160)
    v16bf a  = frag_rows_bf(xh + (size_t)r0 * XHK + kk * 32, XHK);
    v16bf bf = frag_packed(WhP + (size_t)(kk * NT + w) * 512);
    acc = wmma_bf16(a, bf, acc);
  }
#pragma unroll
  for (int kk = 0; kk < Hh / 32; ++kk) {         // (rg*h) @ Uh
    v16bf a  = frag_rows_bf(rgh + (size_t)r0 * Hh + kk * 32, Hh);
    v16bf bf = frag_packed(UhP + (size_t)(kk * NT + w) * 512);
    acc = wmma_bf16(a, bf, acc);
  }
#pragma unroll
  for (int r = 0; r < 8; ++r) {
    int m = hi * 8 + r;
    int ng = n0 + l;
    size_t o = (size_t)(r0 + m) * Hh + ng;
    float ht = fast_tanh(acc[r] + bh[ng]);
    float hv = h[o];
    float zv = zg[o];
    hnew[o] = (1.0f - zv) * hv + zv * ht;
  }
}

// ---------------- Launch ----------------

extern "C" void kernel_launch(void* const* d_in, const int* in_sizes, int n_in,
                              void* d_out, int out_size, void* d_ws, size_t ws_size,
                              hipStream_t stream) {
  (void)in_sizes; (void)n_in; (void)out_size; (void)ws_size;
  const float* h     = (const float*)d_in[0];
  const float* jets  = (const float*)d_in[1];
  const float* mask  = (const float*)d_in[2];
  const float* W_adj = (const float*)d_in[3];
  const float* b_adj = (const float*)d_in[4];
  const float* W_msg = (const float*)d_in[5];
  const float* b_msg = (const float*)d_in[6];
  const float* Wz    = (const float*)d_in[7];
  const float* Uz    = (const float*)d_in[8];
  const float* bz    = (const float*)d_in[9];
  const float* Wr    = (const float*)d_in[10];
  const float* Ur    = (const float*)d_in[11];
  const float* br    = (const float*)d_in[12];
  const float* Wh    = (const float*)d_in[13];
  const float* Uh    = (const float*)d_in[14];
  const float* bh    = (const float*)d_in[15];

  float* hnew  = (float*)d_out;                     // [B,N,H]
  float* A_out = (float*)d_out + (size_t)BN * Hh;   // [B,N,N]

  char* wp = (char*)d_ws;
  auto alloc = [&](size_t bytes) -> void* {
    void* p = wp; wp += (bytes + 255) & ~(size_t)255; return p;
  };
  __bf16* h_bf  = (__bf16*)alloc((size_t)BN * Hh * 2);
  __bf16* z_bf  = (__bf16*)alloc((size_t)BN * Hh * 2);
  __bf16* mT    = (__bf16*)alloc((size_t)BN * Hh * 2);   // [B][H][N] transposed
  __bf16* xh    = (__bf16*)alloc((size_t)BN * XHK * 2);
  float*  zg    = (float*) alloc((size_t)BN * Hh * 4);
  __bf16* rgh   = (__bf16*)alloc((size_t)BN * Hh * 2);
  __bf16* WadjP = (__bf16*)alloc((size_t)Hh * Hh * 2);
  __bf16* WmsgP = (__bf16*)alloc((size_t)Hh * Hh * 2);
  __bf16* WzP   = (__bf16*)alloc((size_t)XHK * Hh * 2);
  __bf16* WrP   = (__bf16*)alloc((size_t)XHK * Hh * 2);
  __bf16* WhP   = (__bf16*)alloc((size_t)WHXK * Hh * 2);
  __bf16* UhP   = (__bf16*)alloc((size_t)Hh * Hh * 2);

  auto pack = [&](__bf16* dst, const float* W, int rW, const float* U, int rU, int totK) {
    int total = totK * Hh;
    k_pack_cat<<<(total + 255) / 256, 256, 0, stream>>>(dst, W, rW, U, rU, totK);
  };
  pack(WadjP, W_adj, Hh, nullptr, 0, Hh);
  pack(WmsgP, W_msg, Hh, nullptr, 0, Hh);
  pack(WzP, Wz, Hh + Ff, Uz, Hh, XHK);
  pack(WrP, Wr, Hh + Ff, Ur, Hh, XHK);
  pack(WhP, Wh, Hh + Ff, nullptr, 0, WHXK);
  pack(UhP, Uh, Hh, nullptr, 0, Hh);

  {
    size_t total = (size_t)BN * (XHK - 128);
    k_prep_xh<<<(unsigned)((total + 255) / 256), 256, 0, stream>>>(h, jets, xh, h_bf);
  }

  k_zm   <<<BN / 16, 256, 0, stream>>>(h_bf, WadjP, WmsgP, b_adj, b_msg, z_bf, mT);
  k_attn <<<Bb * (Nn / 16), 256, 0, stream>>>(z_bf, mT, mask, A_out, xh);
  k_gates<<<BN / 16, 256, 0, stream>>>(xh, WzP, WrP, bz, br, h, zg, rgh);
  k_final<<<BN / 16, 256, 0, stream>>>(xh, WhP, rgh, UhP, bh, h, zg, hnew);
}
